// AttnBlock_62474594287949
// MI455X (gfx1250) — matrix operand order repaired and checked by reading, not compiled, not run on hardware
//
#include <hip/hip_runtime.h>

// ---------------------------------------------------------------------------
// AttnBlock for MI455X (gfx1250): GroupNorm -> QKV proj (WMMA f16 GEMM) ->
// flash attention (WMMA + online softmax) -> out proj + residual (WMMA).
// LDS tiles laid out so WMMA fragments are ds_load_b128; pure-copy stagings
// use GLOBAL_LOAD_ASYNC_TO_LDS_B128 into ping-pong LDS buffers (ASYNCcnt).
// ---------------------------------------------------------------------------

typedef __attribute__((ext_vector_type(16))) _Float16 v16h;
typedef __attribute__((ext_vector_type(8)))  _Float16 v8h;
typedef __attribute__((ext_vector_type(4)))  _Float16 v4h;
typedef __attribute__((ext_vector_type(2)))  _Float16 v2h;
typedef __attribute__((ext_vector_type(8)))  float    v8f;
typedef __attribute__((ext_vector_type(4)))  unsigned int v4u;

#define WMMA(a, b, c) __builtin_amdgcn_wmma_f32_16x16x32_f16( \
    false, (a), false, (b), (short)0, (c), false, false)

#define NB    16
#define NC    512
#define NS    1024
#define NHEAD 8
#define DHEAD 64
#define NGRP  32
#define CPG   16

// async copy of one 16B chunk: global -> LDS, tracked by ASYNCcnt
static __device__ __forceinline__ void async_b128(const _Float16* gptr, _Float16* lptr) {
  const unsigned lds = (unsigned)(unsigned long long)(void*)lptr;
  const unsigned long long ga = (unsigned long long)(const void*)gptr;
  asm volatile("global_load_async_to_lds_b128 %0, %1, off" :: "v"(lds), "v"(ga) : "memory");
}
static __device__ __forceinline__ void wait_async() {
  asm volatile("s_wait_asynccnt 0" ::: "memory");
}

static __device__ __forceinline__ v16h cat8(v8h lo, v8h hi) {
  return __builtin_shufflevector(lo, hi, 0, 1, 2, 3, 4, 5, 6, 7,
                                 8, 9, 10, 11, 12, 13, 14, 15);
}

// A fragment (16x32 f16) from row-major [row][k] tile; elem j -> K =
// 16*(j>>3) + (j&7) + 8*hf  => two contiguous 8-half runs at hf*8 and 16+hf*8.
static __device__ __forceinline__ v16h load_A_frag(const _Float16* base, int ld) {
  const int lane = threadIdx.x & 31;
  const _Float16* p = base + (lane & 15) * ld + (lane >> 4) * 8;
  v8h lo = *(const v8h*)(p);
  v8h hi = *(const v8h*)(p + 16);
  return cat8(lo, hi);
}
// B fragment (32x16 f16) from a [col][k] (transposed) tile; elem j -> K =
// 16*(j>>3) + (j&7) + 8*hf  => two contiguous 8-half runs at hf*8 and 16+hf*8.
static __device__ __forceinline__ v16h load_B_fragT(const _Float16* base, int ld) {
  const int lane = threadIdx.x & 31;
  const _Float16* p = base + (lane & 15) * ld + (lane >> 4) * 8;
  v8h lo = *(const v8h*)(p);
  v8h hi = *(const v8h*)(p + 16);
  return cat8(lo, hi);
}

// row max across the 16 lanes of each half-wave: VALU permlane16 butterflies
static __device__ __forceinline__ float rowmax16(float x) {
#if __has_builtin(__builtin_amdgcn_permlane16)
  unsigned u;
  u = __float_as_uint(x);
  x = fmaxf(x, __uint_as_float(__builtin_amdgcn_permlane16(u, u, 0x67452301u, 0xEFCDAB89u, false, false)));
  u = __float_as_uint(x);
  x = fmaxf(x, __uint_as_float(__builtin_amdgcn_permlane16(u, u, 0x54761032u, 0xDCFE98BAu, false, false)));
  u = __float_as_uint(x);
  x = fmaxf(x, __uint_as_float(__builtin_amdgcn_permlane16(u, u, 0x32107654u, 0xBA98FEDCu, false, false)));
  u = __float_as_uint(x);
  x = fmaxf(x, __uint_as_float(__builtin_amdgcn_permlane16(u, u, 0xFEDCBA98u, 0x76543210u, false, false)));
#elif __has_builtin(__builtin_amdgcn_ds_swizzle)
  x = fmaxf(x, __uint_as_float(__builtin_amdgcn_ds_swizzle(__float_as_uint(x), 0x041f)));
  x = fmaxf(x, __uint_as_float(__builtin_amdgcn_ds_swizzle(__float_as_uint(x), 0x081f)));
  x = fmaxf(x, __uint_as_float(__builtin_amdgcn_ds_swizzle(__float_as_uint(x), 0x101f)));
  x = fmaxf(x, __uint_as_float(__builtin_amdgcn_ds_swizzle(__float_as_uint(x), 0x201f)));
#else
#pragma unroll
  for (int m = 1; m < 16; m <<= 1) x = fmaxf(x, __shfl_xor(x, m, 32));
#endif
  return x;
}

// --- kernel 0: f32 -> f16 weight conversion ---------------------------------
__global__ void convert_w_kernel(const float* __restrict__ w0, const float* __restrict__ w1,
                                 const float* __restrict__ w2, const float* __restrict__ w3,
                                 _Float16* __restrict__ o0, _Float16* __restrict__ o1,
                                 _Float16* __restrict__ o2, _Float16* __restrict__ o3) {
  const float* src = (blockIdx.y == 0) ? w0 : (blockIdx.y == 1) ? w1 : (blockIdx.y == 2) ? w2 : w3;
  _Float16*   dst  = (blockIdx.y == 0) ? o0 : (blockIdx.y == 1) ? o1 : (blockIdx.y == 2) ? o2 : o3;
  const int i = (blockIdx.x * 256 + threadIdx.x) * 4;
  if (i < NC * NC) {
    float4 f = *(const float4*)(src + i);
    v4h h; h.x = (_Float16)f.x; h.y = (_Float16)f.y; h.z = (_Float16)f.z; h.w = (_Float16)f.w;
    *(v4h*)(dst + i) = h;
  }
}

// --- kernel 1: GroupNorm -> f16 [B][C][S] -----------------------------------
__global__ void groupnorm_kernel(const float* __restrict__ x,
                                 const float* __restrict__ gamma,
                                 const float* __restrict__ beta,
                                 _Float16* __restrict__ h16) {
  const int b = blockIdx.x >> 5;
  const int g = blockIdx.x & 31;
  const size_t base = ((size_t)b * NC + g * CPG) * NS;
  const int N = CPG * NS;
  const int tid = threadIdx.x;

  float s = 0.f, ss = 0.f;
#pragma unroll
  for (int j = 0; j < 16; ++j) {
    const int i = (j * 256 + tid) * 4;
    float4 f = *(const float4*)(x + base + i);
    s += f.x + f.y + f.z + f.w;
    ss += f.x * f.x + f.y * f.y + f.z * f.z + f.w * f.w;
  }
  __shared__ float rs[256], rq[256];
  rs[tid] = s; rq[tid] = ss;
  __syncthreads();
#pragma unroll
  for (int off = 128; off > 0; off >>= 1) {
    if (tid < off) { rs[tid] += rs[tid + off]; rq[tid] += rq[tid + off]; }
    __syncthreads();
  }
  const float inv_n = 1.f / (float)N;
  const float mean = rs[0] * inv_n;
  const float var  = rq[0] * inv_n - mean * mean;
  const float rstd = rsqrtf(var + 1e-6f);

#pragma unroll
  for (int j = 0; j < 16; ++j) {
    const int i = (j * 256 + tid) * 4;
    const int c = g * CPG + (i >> 10);
    const float gm = gamma[c] * rstd, bt = beta[c] - mean * gamma[c] * rstd;
    float4 f = *(const float4*)(x + base + i);
    v4h h;
    h.x = (_Float16)(f.x * gm + bt); h.y = (_Float16)(f.y * gm + bt);
    h.z = (_Float16)(f.z * gm + bt); h.w = (_Float16)(f.w * gm + bt);
    *(v4h*)(h16 + base + i) = h;
  }
}

// --- kernels 2/5: out = W(512x512) x X(512x1024) per batch ------------------
// block tile 128x128, K-step 32, ping-pong LDS; A tile via async-to-LDS DMA,
// B tile via register transpose staged after the WMMAs issue.
template <bool FINAL>
__global__ void proj_kernel(const _Float16* __restrict__ W16,
                            const _Float16* __restrict__ X16,
                            const float*    __restrict__ bias,
                            const float*    __restrict__ residual,
                            _Float16*       __restrict__ Y16,
                            float*          __restrict__ Yf) {
  const int m0 = blockIdx.x * 128;
  const int n0 = blockIdx.y * 128;
  const int b  = blockIdx.z;
  const int tid  = threadIdx.x;
  const int wave = tid >> 5;
  const int wm = wave >> 2;
  const int wn = wave & 3;

  __shared__ __align__(16) _Float16 As[2][128][40];   // [m][k]
  __shared__ __align__(16) _Float16 Bs[2][128][40];   // [n][k] (transposed)

  const _Float16* Xb = X16 + (size_t)b * NC * NS;

  v8f acc[4][2];
#pragma unroll
  for (int mi = 0; mi < 4; ++mi)
#pragma unroll
    for (int ni = 0; ni < 2; ++ni)
#pragma unroll
      for (int v = 0; v < 8; ++v) acc[mi][ni][v] = 0.f;

  // A tile: 128x32 halfs = 512 16B chunks, 2 per thread, async DMA to LDS
  auto asyncA = [&](int k0, int buf) {
#pragma unroll
    for (int c = 0; c < 2; ++c) {
      const int chunk = tid + c * 256;
      const int row = chunk >> 2, off = (chunk & 3) * 8;
      async_b128(W16 + (size_t)(m0 + row) * NC + k0 + off, &As[buf][row][off]);
    }
  };
  v2h rb[8];
  auto loadB = [&](int k0) {
#pragma unroll
    for (int j = 0; j < 8; ++j) {
      const int i = tid + j * 256;
      const int kk = i >> 6, n = (i & 63) * 2;
      rb[j] = *(const v2h*)(Xb + (size_t)(k0 + kk) * NS + n0 + n);
    }
  };
  auto storeB = [&](int buf) {
#pragma unroll
    for (int j = 0; j < 8; ++j) {
      const int i = tid + j * 256;
      const int kk = i >> 6, n = (i & 63) * 2;
      Bs[buf][n][kk] = rb[j].x; Bs[buf][n + 1][kk] = rb[j].y;
    }
  };

  asyncA(0, 0);
  loadB(0);
  storeB(0);
  wait_async();
  __syncthreads();

  for (int k0 = 0, it = 0; k0 < NC; k0 += 32, ++it) {
    const int cur = it & 1, nxt = cur ^ 1;
    const bool more = (k0 + 32) < NC;
    if (more) { asyncA(k0 + 32, nxt); loadB(k0 + 32); }

    v16h bf0 = load_B_fragT(&Bs[cur][wn * 32][0], 40);
    v16h bf1 = load_B_fragT(&Bs[cur][wn * 32 + 16][0], 40);
#pragma unroll
    for (int mi = 0; mi < 4; ++mi) {
      v16h af = load_A_frag(&As[cur][wm * 64 + mi * 16][0], 40);
      acc[mi][0] = WMMA(af, bf0, acc[mi][0]);
      acc[mi][1] = WMMA(af, bf1, acc[mi][1]);
    }
    if (more) storeB(nxt);   // overlap global latency with the WMMAs above
    wait_async();
    __syncthreads();
  }

  const int lane = tid & 31, col = lane & 15, hf = lane >> 4;
#pragma unroll
  for (int mi = 0; mi < 4; ++mi) {
#pragma unroll
    for (int ni = 0; ni < 2; ++ni) {
#pragma unroll
      for (int v = 0; v < 8; ++v) {
        const int m = m0 + wm * 64 + mi * 16 + v + 8 * hf;
        const int n = n0 + wn * 32 + ni * 16 + col;
        const size_t idx = ((size_t)b * NC + m) * NS + n;
        const float val = acc[mi][ni][v] + bias[m];
        if (FINAL) Yf[idx] = residual[idx] + val;
        else       Y16[idx] = (_Float16)val;
      }
    }
  }
}

// --- kernel 3: flash attention ---------------------------------------------
// grid (S/128, B*NHEAD); 8 waves x 16 query rows; 32-key chunks; ping-pong
// LDS; V tile via async-to-LDS DMA, K tile via register transpose.
__global__ void attn_kernel(const _Float16* __restrict__ q16,
                            const _Float16* __restrict__ k16,
                            const _Float16* __restrict__ v16p,
                            _Float16*       __restrict__ o16,
                            float scale) {
  const int qt = blockIdx.x;
  const int bh = blockIdx.y;
  const int b  = bh >> 3;
  const int h  = bh & 7;
  const size_t base = ((size_t)b * NC + h * DHEAD) * NS;

  __shared__ __align__(16) union {
    _Float16 Q[128][72];   // [s][dd]  A-operand of Q*K^T
    _Float16 O[64][136];   // [dd][s]  output staging
  } QO;
  __shared__ __align__(16) _Float16 Ks[2][32][72];    // [key][dd]  B of Q*K^T
  __shared__ __align__(16) _Float16 Vs[2][64][40];    // [dd][key]  B of P*V
  __shared__ __align__(16) _Float16 Ps[8][16][40];    // per-wave P (A layout)

  const int tid  = threadIdx.x;
  const int wave = tid >> 5;
  const int lane = tid & 31;
  const int col  = lane & 15;
  const int hf   = lane >> 4;

  // stage Q: coalesced along s, transpose into [s][dd]
#pragma unroll
  for (int j = 0; j < 16; ++j) {
    const int i = tid + j * 256;
    const int dd = i >> 6, s = (i & 63) * 2;
    v2h p = *(const v2h*)(q16 + base + (size_t)dd * NS + qt * 128 + s);
    QO.Q[s][dd] = p.x; QO.Q[s + 1][dd] = p.y;
  }

  // V chunk: 64x32 halfs = 256 16B chunks, 1 per thread, async DMA to LDS
  auto asyncV = [&](int j0, int buf) {
    const int dd = tid >> 2, off = (tid & 3) * 8;
    async_b128(v16p + base + (size_t)dd * NS + j0 + off, &Vs[buf][dd][off]);
  };
  v2h rk[4];
  auto loadK = [&](int j0) {
#pragma unroll
    for (int j = 0; j < 4; ++j) {
      const int i = tid + j * 256;
      const int dd = i >> 4, kk = (i & 15) * 2;
      rk[j] = *(const v2h*)(k16 + base + (size_t)dd * NS + j0 + kk);
    }
  };
  auto storeK = [&](int buf) {
#pragma unroll
    for (int j = 0; j < 4; ++j) {
      const int i = tid + j * 256;
      const int dd = i >> 4, kk = (i & 15) * 2;
      Ks[buf][kk][dd] = rk[j].x; Ks[buf][kk + 1][dd] = rk[j].y;
    }
  };

  asyncV(0, 0);
  loadK(0);
  storeK(0);
  wait_async();
  __syncthreads();

  const v16h qf0 = load_A_frag(&QO.Q[wave * 16][0], 72);
  const v16h qf1 = load_A_frag(&QO.Q[wave * 16][32], 72);

  v16h onef;
#pragma unroll
  for (int j = 0; j < 16; ++j) onef[j] = (_Float16)1.0f;

  v8f o_acc[4];
#pragma unroll
  for (int t = 0; t < 4; ++t)
#pragma unroll
    for (int v = 0; v < 8; ++v) o_acc[t][v] = 0.f;
  float m_run[8], l_run[8];
#pragma unroll
  for (int v = 0; v < 8; ++v) { m_run[v] = -1e30f; l_run[v] = 0.f; }

  for (int j0 = 0, it = 0; j0 < NS; j0 += 32, ++it) {
    const int cur = it & 1, nxt = cur ^ 1;
    const bool more = (j0 + 32) < NS;
    if (more) { asyncV(j0 + 32, nxt); loadK(j0 + 32); }

    // scores: 2 tiles of 16x16 over 32 keys, K-dim = 64
    v8f s0, s1;
#pragma unroll
    for (int v = 0; v < 8; ++v) { s0[v] = 0.f; s1[v] = 0.f; }
    {
      v16h b00 = load_B_fragT(&Ks[cur][0][0], 72);
      v16h b01 = load_B_fragT(&Ks[cur][0][32], 72);
      v16h b10 = load_B_fragT(&Ks[cur][16][0], 72);
      v16h b11 = load_B_fragT(&Ks[cur][16][32], 72);
      s0 = WMMA(qf0, b00, s0); s0 = WMMA(qf1, b01, s0);
      s1 = WMMA(qf0, b10, s1); s1 = WMMA(qf1, b11, s1);
    }

    // online softmax: row max via permlane butterflies
    float alpha[8];
#pragma unroll
    for (int v = 0; v < 8; ++v) {
      const float a0 = s0[v] * scale;
      const float a1 = s1[v] * scale;
      const float mx = rowmax16(fmaxf(a0, a1));
      const float mn = fmaxf(m_run[v], mx);
      alpha[v] = __expf(m_run[v] - mn);
      m_run[v] = mn;
      s0[v] = __expf(a0 - mn);
      s1[v] = __expf(a1 - mn);
    }

    // P -> A-fragment layout via per-wave LDS region
#pragma unroll
    for (int v = 0; v < 8; ++v) {
      const int r = v + 8 * hf;
      Ps[wave][r][col]      = (_Float16)s0[v];
      Ps[wave][r][col + 16] = (_Float16)s1[v];
    }
    const v16h pf = load_A_frag(&Ps[wave][0][0], 40);

    // row sums of P via WMMA against all-ones B
    v8f rsum;
#pragma unroll
    for (int v = 0; v < 8; ++v) rsum[v] = 0.f;
    rsum = WMMA(pf, onef, rsum);

#pragma unroll
    for (int v = 0; v < 8; ++v) l_run[v] = l_run[v] * alpha[v] + rsum[v];
#pragma unroll
    for (int t = 0; t < 4; ++t)
#pragma unroll
      for (int v = 0; v < 8; ++v) o_acc[t][v] = o_acc[t][v] * alpha[v];

    // O += P(16x32) x V(32x64)
    v16h vb0 = load_B_fragT(&Vs[cur][0][0], 40);
    v16h vb1 = load_B_fragT(&Vs[cur][16][0], 40);
    v16h vb2 = load_B_fragT(&Vs[cur][32][0], 40);
    v16h vb3 = load_B_fragT(&Vs[cur][48][0], 40);
    o_acc[0] = WMMA(pf, vb0, o_acc[0]);
    o_acc[1] = WMMA(pf, vb1, o_acc[1]);
    o_acc[2] = WMMA(pf, vb2, o_acc[2]);
    o_acc[3] = WMMA(pf, vb3, o_acc[3]);

    if (more) storeK(nxt);   // after WMMAs: overlap global latency
    wait_async();
    __syncthreads();
  }

  // normalize, stage through LDS ([dd][s]) for coalesced b128 stores
  float inv_l[8];
#pragma unroll
  for (int v = 0; v < 8; ++v) inv_l[v] = 1.0f / l_run[v];
#pragma unroll
  for (int t = 0; t < 4; ++t)
#pragma unroll
    for (int v = 0; v < 8; ++v) {
      const int r = v + 8 * hf;
      QO.O[t * 16 + col][wave * 16 + r] = (_Float16)(o_acc[t][v] * inv_l[v]);
    }
  __syncthreads();
#pragma unroll
  for (int j = 0; j < 4; ++j) {
    const int c = tid + j * 256;
    const int dd = c >> 4, s0i = (c & 15) * 8;
    *(v4u*)(o16 + base + (size_t)dd * NS + qt * 128 + s0i) = *(const v4u*)&QO.O[dd][s0i];
  }
}

// ---------------------------------------------------------------------------
extern "C" void kernel_launch(void* const* d_in, const int* in_sizes, int n_in,
                              void* d_out, int out_size, void* d_ws, size_t ws_size,
                              hipStream_t stream) {
  const float* x     = (const float*)d_in[0];
  const float* gamma = (const float*)d_in[1];
  const float* beta  = (const float*)d_in[2];
  const float* wq    = (const float*)d_in[3];
  const float* bq    = (const float*)d_in[4];
  const float* wk    = (const float*)d_in[5];
  const float* bk    = (const float*)d_in[6];
  const float* wv    = (const float*)d_in[7];
  const float* bv    = (const float*)d_in[8];
  const float* wo    = (const float*)d_in[9];
  const float* bo    = (const float*)d_in[10];
  float* out = (float*)d_out;

  const size_t wsz = (size_t)NC * NC;
  const size_t tsz = (size_t)NB * NC * NS;
  char* ws = (char*)d_ws;
  _Float16* w16q = (_Float16*)ws;
  _Float16* w16k = w16q + wsz;
  _Float16* w16v = w16k + wsz;
  _Float16* w16o = w16v + wsz;
  _Float16* h16  = w16o + wsz;
  _Float16* q16  = h16 + tsz;
  _Float16* k16  = q16 + tsz;
  _Float16* v16  = k16 + tsz;
  _Float16* o16  = v16 + tsz;

  convert_w_kernel<<<dim3(NC * NC / (256 * 4), 4), 256, 0, stream>>>(
      wq, wk, wv, wo, w16q, w16k, w16v, w16o);

  groupnorm_kernel<<<NB * NGRP, 256, 0, stream>>>(x, gamma, beta, h16);

  const dim3 pg(NC / 128, NS / 128, NB);
  proj_kernel<false><<<pg, 256, 0, stream>>>(w16q, h16, bq, nullptr, q16, nullptr);
  proj_kernel<false><<<pg, 256, 0, stream>>>(w16k, h16, bk, nullptr, k16, nullptr);
  proj_kernel<false><<<pg, 256, 0, stream>>>(w16v, h16, bv, nullptr, v16, nullptr);

  const float scale = 0.044194173824159216f;  // C^-0.5 (full C, as in source)
  attn_kernel<<<dim3(NS / 128, NB * NHEAD), 256, 0, stream>>>(q16, k16, v16, o16, scale);

  proj_kernel<true><<<pg, 256, 0, stream>>>(w16o, o16, bo, x, nullptr, out);
}
